// BiLSTM_CRF_6296422056097
// MI455X (gfx1250) — matrix operand order repaired and checked by reading, not compiled, not run on hardware
//
#include <hip/hip_runtime.h>
#include <hip/hip_bf16.h>
#include <math.h>

// BiLSTM-CRF on MI455X (gfx1250, wave32, WMMA).
// Fused recurrent step: 4 waves/block compute the i,f,g,o 16x16 gate tiles of
// one (batch-tile, hidden-tile) pair with v_wmma_f32_16x16x32_f16, stage them
// in LDS, then do the cell/hidden update in the same launch. h is ping-pong
// double-buffered across launches (launch boundary = inter-step sync).

typedef __attribute__((ext_vector_type(16))) _Float16 v16h;
typedef __attribute__((ext_vector_type(8)))  _Float16 v8h;
typedef __attribute__((ext_vector_type(8)))  float    v8f;

#define B_   64
#define T_   512
#define E_   512
#define H_   512
#define G4_  2048      // 4*H (gate dim, torch order i,f,g,o)
#define KC_  1024      // concat K = E + H
#define KT_  12        // tags
#define KP_  16        // padded tag dim for WMMA N
#define STARTT 10
#define STOPT  11
#define NEGV  (-10000.0f)

// ---------------- init kernels ----------------

// Fuse [Wi ; Wh] -> Wcat f16, row-major per gate row n (contiguous K).
__global__ void k_wcat(const float* __restrict__ Wi_f, const float* __restrict__ Wh_f,
                       const float* __restrict__ Wi_b, const float* __restrict__ Wh_b,
                       _Float16* __restrict__ Wcat) {
  size_t i = (size_t)blockIdx.x * blockDim.x + threadIdx.x;   // 2*G4_*KC_
  if (i >= (size_t)2 * G4_ * KC_) return;
  int k = (int)(i % KC_);
  size_t r = i / KC_;
  int n = (int)(r % G4_);
  int dir = (int)(r / G4_);
  const float* Wi = dir ? Wi_b : Wi_f;
  const float* Wh = dir ? Wh_b : Wh_f;
  float v = (k < E_) ? Wi[(size_t)n * E_ + k] : Wh[(size_t)n * H_ + (k - E_)];
  Wcat[i] = (_Float16)v;
}

// W_out^T padded to 16 columns, stored column-major (per n contiguous K).
__global__ void k_wout(const float* __restrict__ W_out, _Float16* __restrict__ Wout16) {
  int i = blockIdx.x * blockDim.x + threadIdx.x;   // KP_*KC_
  if (i >= KP_ * KC_) return;
  int k = i % KC_;
  int n = i / KC_;
  float v = (n < KT_) ? W_out[(size_t)n * KC_ + k] : 0.f;
  Wout16[i] = (_Float16)v;
}

// h0 -> f16 state (ping buffer), c0 -> f32 state.
__global__ void k_state(const float* __restrict__ h0, const float* __restrict__ c0,
                        _Float16* __restrict__ h16, float* __restrict__ cbuf) {
  int i = blockIdx.x * blockDim.x + threadIdx.x;   // 2*B_*H_
  if (i >= 2 * B_ * H_) return;
  h16[i] = (_Float16)h0[i];
  cbuf[i] = c0[i];
}

// Embedding gather (+mask, +padding_idx 0) -> x_f f16; per-sequence reversal -> x_b f16.
__global__ void k_embed(const int* __restrict__ sentence, const int* __restrict__ seq_lens,
                        const float* __restrict__ embed,
                        _Float16* __restrict__ x_f, _Float16* __restrict__ x_b) {
  size_t i = (size_t)blockIdx.x * blockDim.x + threadIdx.x;   // B_*T_*E_
  int e = (int)(i % E_);
  size_t r = i / E_;
  int t = (int)(r % T_);
  int b = (int)(r / T_);
  int len = seq_lens[b];
  int w = sentence[b * T_ + t];
  float v = 0.f;
  if (t < len && w != 0) v = embed[(size_t)w * E_ + e];
  _Float16 hv = (_Float16)v;
  x_f[i] = hv;
  if (t < len) x_b[((size_t)b * T_ + (len - 1 - t)) * E_ + e] = hv;
  else         x_b[i] = (_Float16)0.f;
}

// ---------------- fused recurrent step ----------------
// grid = 2 dirs * 4 Mtiles * 32 Jtiles = 256 blocks of 128 threads (4 waves).
// Wave g computes gate g's 16x16 tile: gates = [x_t,h] @ Wcat^T + bias (bias
// folded into the WMMA C accumulator: each lane's 8 f32s share one column).
__global__ __launch_bounds__(128)
void k_step(const _Float16* __restrict__ x_f, const _Float16* __restrict__ x_b,
            const _Float16* __restrict__ hin, _Float16* __restrict__ hout,
            const _Float16* __restrict__ Wcat,
            const float* __restrict__ b_f, const float* __restrict__ b_b,
            float* __restrict__ cbuf,
            _Float16* __restrict__ out_f, _Float16* __restrict__ out_b,
            const int* __restrict__ seq_lens, int t) {
  __shared__ float glds[4][16][17];   // +1 pad: half-wave row offset 8*17 != 0 mod 64 banks
  int bid = blockIdx.x;
  int dir = bid >> 7;
  int mt  = (bid >> 5) & 3;
  int jt  = bid & 31;
  int g    = threadIdx.x >> 5;        // gate index == wave id (i,f,g,o)
  int lane = threadIdx.x & 31;
  int lr = lane & 15;
  int sel = lane >> 4;                // half-wave select per ISA A/B layouts
  int m  = mt * 16 + lr;              // batch row
  int n0 = g * H_ + jt * 16;          // gate column base

  const _Float16* x    = dir ? x_b : x_f;
  const _Float16* xrow = x + ((size_t)m * T_ + t) * E_;
  const _Float16* hrow = hin + ((size_t)dir * B_ + m) * H_;
  const _Float16* bcol = Wcat + ((size_t)dir * G4_ + n0 + lr) * KC_;
  const float* bias = dir ? b_b : b_f;

  __builtin_prefetch((const void*)hrow, 0, 0);   // global_prefetch_b8: h used in 2nd half of K loop

  float bv = bias[n0 + lr];
  v8f acc = {bv, bv, bv, bv, bv, bv, bv, bv};
#pragma unroll
  for (int kk = 0; kk < 32; ++kk) {
    int kbase = kk * 32;
    const _Float16* ar = (kbase < E_) ? (xrow + kbase) : (hrow + (kbase - E_));
    v8h alo = *(const v8h*)(ar + sel * 16);
    v8h ahi = *(const v8h*)(ar + 8 + sel * 16);
    v16h A = __builtin_shufflevector(alo, ahi, 0,1,2,3,4,5,6,7,8,9,10,11,12,13,14,15);
    v16h Bm = *(const v16h*)(bcol + kbase + sel * 16);
    acc = __builtin_amdgcn_wmma_f32_16x16x32_f16(false, A, false, Bm,
                                                 (short)0, acc, false, false);
  }
#pragma unroll
  for (int r = 0; r < 8; ++r) glds[g][sel * 8 + r][lr] = acc[r];
  __syncthreads();

  // Cell/hidden update straight out of LDS: 128 threads cover 16x16 elements x2.
  for (int p = threadIdx.x; p < 256; p += 128) {
    int mr = p >> 4;
    int jc = p & 15;
    int b = mt * 16 + mr;
    int j = jt * 16 + jc;
    float gi = glds[0][mr][jc];
    float gf = glds[1][mr][jc];
    float gg = glds[2][mr][jc];
    float go = glds[3][mr][jc];
    float si = 1.f / (1.f + __expf(-gi));
    float sf = 1.f / (1.f + __expf(-gf));
    float so = 1.f / (1.f + __expf(-go));
    size_t ci = (size_t)(dir * B_ + b) * H_ + j;
    float c = sf * cbuf[ci] + si * tanhf(gg);
    cbuf[ci] = c;
    float h = so * tanhf(c);
    hout[ci] = (_Float16)h;
    if (dir == 0) {
      out_f[((size_t)b * T_ + t) * H_ + j] = (_Float16)h;
    } else {
      int len = seq_lens[b];
      if (t < len) out_b[((size_t)b * T_ + (len - 1 - t)) * H_ + j] = (_Float16)h;
    }
  }
}

// ---------------- output projection: feats = [out_f,out_b] @ W_out^T + b ----------------
__global__ __launch_bounds__(32)
void k_proj(const _Float16* __restrict__ out_f, const _Float16* __restrict__ out_b,
            const _Float16* __restrict__ Wout16, const float* __restrict__ b_out,
            float* __restrict__ feats) {
  int m0 = blockIdx.x * 16;       // 2048 blocks cover B*T rows
  int lane = threadIdx.x;
  int lr = lane & 15;
  int sel = lane >> 4;
  int m = m0 + lr;
  const _Float16* fr = out_f + (size_t)m * H_;
  const _Float16* br = out_b + (size_t)m * H_;
  const _Float16* bcol = Wout16 + (size_t)lr * KC_;
  v8f acc = {};
#pragma unroll
  for (int kk = 0; kk < 32; ++kk) {
    int kbase = kk * 32;
    const _Float16* ar = (kbase < H_) ? (fr + kbase) : (br + (kbase - H_));
    v8h alo = *(const v8h*)(ar + sel * 16);
    v8h ahi = *(const v8h*)(ar + 8 + sel * 16);
    v16h A = __builtin_shufflevector(alo, ahi, 0,1,2,3,4,5,6,7,8,9,10,11,12,13,14,15);
    v16h Bm = *(const v16h*)(bcol + kbase + sel * 16);
    acc = __builtin_amdgcn_wmma_f32_16x16x32_f16(false, A, false, Bm,
                                                 (short)0, acc, false, false);
  }
  float bb = (lr < KT_) ? b_out[lr] : 0.f;
  float* fo = feats + (size_t)(m0 + sel * 8) * KP_ + lr;
#pragma unroll
  for (int r = 0; r < 8; ++r) fo[(size_t)r * KP_] = acc[r] + bb;
}

// ---------------- Viterbi forward + backtrack (one wave per batch row) ----------------
__global__ void k_viterbi(const float* __restrict__ feats, const int* __restrict__ seq_lens,
                          const float* __restrict__ trans, unsigned char* __restrict__ bps,
                          float* __restrict__ out) {
  int b = blockIdx.x;
  int tid = threadIdx.x;
  __shared__ float tr[KT_ * KT_];
  __shared__ float fv[KT_], fvn[KT_];
  for (int i = tid; i < KT_ * KT_; i += 32) tr[i] = trans[i];
  if (tid < KT_) fv[tid] = (tid == STARTT) ? 0.f : NEGV;
  __syncthreads();
  int len = seq_lens[b];
  for (int t = 0; t < T_; ++t) {
    if (tid < KT_) {
      float best = -3.4e38f; int arg = 0;
#pragma unroll
      for (int p = 0; p < KT_; ++p) {
        float s = fv[p] + tr[tid * KT_ + p];
        if (s > best) { best = s; arg = p; }
      }
      float val = best + feats[((size_t)b * T_ + t) * KP_ + tid];
      size_t bi = ((size_t)b * T_ + t) * KT_ + tid;
      if (t < len) { fvn[tid] = val;     bps[bi] = (unsigned char)arg; }
      else         { fvn[tid] = fv[tid]; bps[bi] = (unsigned char)tid; }
    }
    __syncthreads();
    if (tid < KT_) fv[tid] = fvn[tid];
    __syncthreads();
  }
  if (tid == 0) {
    float best = -3.4e38f; int tag = 0;
    for (int p = 0; p < KT_; ++p) {
      float s = fv[p] + tr[STOPT * KT_ + p];
      if (s > best) { best = s; tag = p; }
    }
    out[b] = best;                                // scores
    for (int t = T_ - 1; t >= 0; --t) {
      out[B_ + (size_t)b * T_ + t] = (float)tag;  // preds
      tag = bps[((size_t)b * T_ + t) * KT_ + tag];
    }
  }
}

// ---------------- host launcher ----------------
extern "C" void kernel_launch(void* const* d_in, const int* in_sizes, int n_in,
                              void* d_out, int out_size, void* d_ws, size_t ws_size,
                              hipStream_t stream) {
  (void)in_sizes; (void)n_in; (void)out_size; (void)ws_size;
  const int*   sentence = (const int*)d_in[0];
  const int*   seq_lens = (const int*)d_in[1];
  const float* embed    = (const float*)d_in[2];
  const float* Wi_f = (const float*)d_in[3];
  const float* Wh_f = (const float*)d_in[4];
  const float* b_f  = (const float*)d_in[5];
  const float* Wi_b = (const float*)d_in[6];
  const float* Wh_b = (const float*)d_in[7];
  const float* b_b  = (const float*)d_in[8];
  const float* h0   = (const float*)d_in[9];
  const float* c0   = (const float*)d_in[10];
  const float* W_out = (const float*)d_in[11];
  const float* b_out = (const float*)d_in[12];
  const float* trans = (const float*)d_in[13];
  float* out = (float*)d_out;

  char* ws = (char*)d_ws;
  size_t off = 0;
  auto alloc = [&](size_t bytes) -> void* {
    void* p = ws + off;
    off = (off + bytes + 255) & ~(size_t)255;
    return p;
  };
  _Float16* x_f    = (_Float16*)alloc((size_t)B_ * T_ * E_ * 2);
  _Float16* x_b    = (_Float16*)alloc((size_t)B_ * T_ * E_ * 2);
  _Float16* Wcat   = (_Float16*)alloc((size_t)2 * G4_ * KC_ * 2);
  _Float16* Wout16 = (_Float16*)alloc((size_t)KP_ * KC_ * 2);
  _Float16* h_ping = (_Float16*)alloc((size_t)2 * B_ * H_ * 2);
  _Float16* h_pong = (_Float16*)alloc((size_t)2 * B_ * H_ * 2);
  float*    cbuf   = (float*)   alloc((size_t)2 * B_ * H_ * 4);
  _Float16* out_f  = (_Float16*)alloc((size_t)B_ * T_ * H_ * 2);
  _Float16* out_b  = (_Float16*)alloc((size_t)B_ * T_ * H_ * 2);
  float*    feats  = (float*)   alloc((size_t)B_ * T_ * KP_ * 4);
  unsigned char* bps = (unsigned char*)alloc((size_t)B_ * T_ * KT_);

  // init
  k_wcat <<<(2 * G4_ * KC_ + 255) / 256, 256, 0, stream>>>(Wi_f, Wh_f, Wi_b, Wh_b, Wcat);
  k_wout <<<(KP_ * KC_ + 255) / 256,      256, 0, stream>>>(W_out, Wout16);
  k_state<<<(2 * B_ * H_ + 255) / 256,    256, 0, stream>>>(h0, c0, h_ping, cbuf);
  k_embed<<<((size_t)B_ * T_ * E_ + 255) / 256, 256, 0, stream>>>(sentence, seq_lens, embed, x_f, x_b);

  // recurrent chain: one fused launch per step, h ping-pong double-buffered
  for (int t = 0; t < T_; ++t) {
    const _Float16* hin = (t & 1) ? h_pong : h_ping;
    _Float16*      hout = (t & 1) ? h_ping : h_pong;
    k_step<<<256, 128, 0, stream>>>(x_f, x_b, hin, hout, Wcat, b_f, b_b, cbuf,
                                    out_f, out_b, seq_lens, t);
  }

  // projection to tag features, then Viterbi
  k_proj   <<<(B_ * T_) / 16, 32, 0, stream>>>(out_f, out_b, Wout16, b_out, feats);
  k_viterbi<<<B_, 32, 0, stream>>>(feats, seq_lens, trans, bps, out);
}
